// ETNNLayer_88622355186349
// MI455X (gfx1250) — matrix operand order repaired and checked by reading, not compiled, not run on hardware
//
#include <hip/hip_runtime.h>
#include <hip/hip_bf16.h>
#include <stdint.h>

// ---------------------------------------------------------------------------
// ETNN layer, restructured:
//   prep:   cent = mean of 4 owned nodes; U[:,0:64] = (f16)features
//   pack:   weights -> f16, rearranged into WMMA B-fragment lane layout
//   G1:     X1 = U[:,0:64] @ [W1a | W1b]            (f32 out, N x 128)
//   pair:   S  = (1/16) sum_j silu(A_i + B_j + g_ij*w1g + b1)   (f16)
//   G2:     U[:,64:128] = (f16)(S @ W2 + b2)        (= msg)
//   G3:     HU = (f16) silu(U @ Wu1 + bu1)          (K = 128)
//   G4:     d_out = features + HU @ Wu2 + bu2       (f32, residual fused)
//   positions copied through unchanged.
// All matmuls use v_wmma_f32_16x16x32_f16 (f16 in, f32 accumulate).
// N = 50000 is divisible by 16 -> every GEMM block covers exactly 16 rows,
// so there are NO bounds checks (branch-free epilogue, EXEC always full).
// ---------------------------------------------------------------------------

typedef __attribute__((ext_vector_type(16))) _Float16 v16h;
typedef __attribute__((ext_vector_type(8)))  float    v8f;
typedef unsigned int u32;

// ---------------- prep: features -> f16 into U[:,0:64], centroids ----------
__global__ __launch_bounds__(256)
void prep_kernel(const float* __restrict__ feat, const float* __restrict__ pos,
                 _Float16* __restrict__ U, float* __restrict__ cent4, int N)
{
    int t = blockIdx.x * blockDim.x + threadIdx.x;
    if (t < N * 64) {
        int i = t >> 6, ch = t & 63;
        U[(size_t)i * 128 + ch] = (_Float16)feat[t];
    }
    if (t < N * 3) {
        int i = t / 3, c = t % 3;
        float s = pos[(size_t)(4 * i + 0) * 3 + c] + pos[(size_t)(4 * i + 1) * 3 + c]
                + pos[(size_t)(4 * i + 2) * 3 + c] + pos[(size_t)(4 * i + 3) * 3 + c];
        cent4[(size_t)i * 4 + c] = 0.25f * s;
    }
}

// ---------------- weight packing into B-fragment layout --------------------
// B is logically [K x Nn] (K mult of 32, Nn mult of 16). Packed u32 index:
//   idx = ((kt*NT + nt)*32 + lane)*8 + v
//   lane: half = lane>>4, r = lane&15;  col = nt*16 + r
//   VGPR v holds packed halves for K rows (kt*32 + half*8 + 2v + (v>=4 ? 8 : 0), +1)
// mode 0: src is row-major [K x Nn]
// mode 1: W1 split: src is [129 x 64]; B[k][n] = n<64 ? W1[k][n] : W1[64+k][n-64]
__global__ __launch_bounds__(256)
void pack_b_kernel(const float* __restrict__ src, u32* __restrict__ dst,
                   int K, int Nn, int mode)
{
    int idx = blockIdx.x * blockDim.x + threadIdx.x;
    int total = (K / 32) * (Nn / 16) * 256;
    if (idx >= total) return;
    int v    = idx & 7;
    int lane = (idx >> 3) & 31;
    int tile = idx >> 8;
    int NT = Nn / 16;
    int kt = tile / NT, nt = tile % NT;
    int half_ = lane >> 4, r = lane & 15;
    int k0  = kt * 32 + half_ * 8 + 2 * v + (v >= 4 ? 8 : 0);
    int col = nt * 16 + r;
    float f0, f1;
    if (mode == 0) {
        f0 = src[(size_t)k0 * Nn + col];
        f1 = src[(size_t)(k0 + 1) * Nn + col];
    } else {
        if (col < 64) {
            f0 = src[(size_t)k0 * 64 + col];
            f1 = src[(size_t)(k0 + 1) * 64 + col];
        } else {
            f0 = src[(size_t)(64 + k0) * 64 + (col - 64)];
            f1 = src[(size_t)(64 + k0 + 1) * 64 + (col - 64)];
        }
    }
    union { _Float16 h[2]; u32 u; } p;
    p.h[0] = (_Float16)f0;
    p.h[1] = (_Float16)f1;
    dst[idx] = p.u;
}

// ---------------- WMMA GEMM: D[M x (NT*16)] = A[M x K] * B + epilogue ------
// One wave computes one 16x16 tile; NT waves per block; every block covers
// exactly 16 rows (requires M % 16 == 0 -> no bounds checks anywhere).
// LDA/LDD are compile-time so all addressing is shift+immediate-offset.
template<int KTILES, int NT, int LDA, int LDD, bool SILU, bool HALF_OUT,
         bool RESID, bool BIAS>
__global__ __launch_bounds__(NT * 32)
void gemm_wmma_kernel(const _Float16* __restrict__ A,
                      const u32* __restrict__ Bp,
                      const float* __restrict__ bias,
                      const float* __restrict__ resid,
                      void* __restrict__ Dst)
{
    const int lane  = threadIdx.x & 31;
    const int nt    = threadIdx.x >> 5;          // wave id == column tile
    const int m0    = blockIdx.x * 16;
    const int half_ = lane >> 4;
    const int r     = lane & 15;

    const u32* ap = (const u32*)(A + (size_t)(m0 + r) * LDA);

    v8f c = {0.f, 0.f, 0.f, 0.f, 0.f, 0.f, 0.f, 0.f};

#pragma unroll
    for (int kt = 0; kt < KTILES; ++kt) {
        const int k0 = kt * 32;
        union { v16h v; uint4 q[2]; } af, bf;
        // A fragment: lanes 0-15 = rows, halves K [k0+half*8 .. +7] and [k0+16+half*8 .. +7]
        af.q[0] = *(const uint4*)(ap + ((k0 + half_ * 8) >> 1));
        af.q[1] = *(const uint4*)(ap + ((k0 + 16 + half_ * 8) >> 1));
        // B fragment: pre-packed, two b128 per lane
        const uint4* bp4 = (const uint4*)(Bp + (((size_t)(kt * NT + nt)) * 32 + lane) * 8);
        bf.q[0] = bp4[0];
        bf.q[1] = bp4[1];
        c = __builtin_amdgcn_wmma_f32_16x16x32_f16(false, af.v, false, bf.v,
                                                   (short)0, c, false, false);
    }

    const int ocol  = nt * 16 + r;
    const int orow0 = m0 + half_ * 8;            // this lane owns rows orow0..orow0+7
    float bv = 0.f;
    if (BIAS) bv = bias[ocol];
    const float* rp = RESID ? (resid + (size_t)orow0 * 64 + ocol) : nullptr;
    _Float16* dh = (_Float16*)Dst + (size_t)orow0 * LDD + ocol;
    float*    df = (float*)Dst    + (size_t)orow0 * LDD + ocol;

#pragma unroll
    for (int v = 0; v < 8; ++v) {
        float val = c[v];
        if (BIAS)  val += bv;
        if (SILU)  val = val / (1.f + __expf(-val));
        if (RESID) val += rp[v * 64];
        if (HALF_OUT) dh[v * LDD] = (_Float16)val;
        else          df[v * LDD] = val;
    }
}

// ---------------- pair pass: gather + geom + SiLU + mean over neighbors ----
// 256 threads = 4 cells x 64 channels.  N % 4 == 0 (50000/4 = 12500 blocks).
__global__ __launch_bounds__(256)
void pair_pass_kernel(const float* __restrict__ X1, const float* __restrict__ cent4,
                      const int* __restrict__ nbr, const float* __restrict__ b1,
                      const float* __restrict__ w1g, _Float16* __restrict__ S16, int N)
{
    __shared__ float gs[4][16];
    __shared__ int   js[4][16];
    const int t    = threadIdx.x;
    const int g    = t >> 6;
    const int ch   = t & 63;
    const int cell = blockIdx.x * 4 + g;

    if (ch < 16) {
        int j = nbr[(size_t)cell * 16 + ch];
        js[g][ch] = j;
        float dx = cent4[(size_t)cell * 4 + 0] - cent4[(size_t)j * 4 + 0];
        float dy = cent4[(size_t)cell * 4 + 1] - cent4[(size_t)j * 4 + 1];
        float dz = cent4[(size_t)cell * 4 + 2] - cent4[(size_t)j * 4 + 2];
        gs[g][ch] = sqrtf(dx * dx + dy * dy + dz * dz);
    }
    __syncthreads();

    const float ai = X1[(size_t)cell * 128 + ch] + b1[ch];
    const float wg = w1g[ch];
    float acc = 0.f;
#pragma unroll
    for (int j = 0; j < 16; ++j) {
        const int jn = js[g][j];
        float x = ai + X1[(size_t)jn * 128 + 64 + ch] + gs[g][j] * wg;
        acc += x / (1.f + __expf(-x));           // SiLU
    }
    S16[(size_t)cell * 64 + ch] = (_Float16)(acc * (1.f / 16.f));
}

// ---------------------------------------------------------------------------
extern "C" void kernel_launch(void* const* d_in, const int* in_sizes, int n_in,
                              void* d_out, int out_size, void* d_ws, size_t ws_size,
                              hipStream_t stream)
{
    const float* feat = (const float*)d_in[0];   // [N,64]
    const float* pos  = (const float*)d_in[1];   // [4N,3]
    const int*   nbr  = (const int*)  d_in[2];   // [N,16]
    const float* W1   = (const float*)d_in[3];   // [129,64]
    const float* b1   = (const float*)d_in[4];
    const float* W2   = (const float*)d_in[5];   // [64,64]
    const float* b2   = (const float*)d_in[6];
    const float* Wu1  = (const float*)d_in[7];   // [128,64]
    const float* bu1  = (const float*)d_in[8];
    const float* Wu2  = (const float*)d_in[9];   // [64,64]
    const float* bu2  = (const float*)d_in[10];
    const int N = in_sizes[0] / 64;              // 50000 (multiple of 16)

    // workspace carve-up (~52 MB total)
    char* ws = (char*)d_ws;
    size_t off = 0;
    auto carve = [&](size_t bytes) -> void* {
        void* p = ws + off;
        off = (off + bytes + 255) & ~(size_t)255;
        return p;
    };
    _Float16* U    = (_Float16*)carve((size_t)N * 128 * sizeof(_Float16)); // [f16 feat | f16 msg]
    float*    X1   = (float*)   carve((size_t)N * 128 * sizeof(float));    // [A | B] projections
    _Float16* S16  = (_Float16*)carve((size_t)N * 64  * sizeof(_Float16)); // mean-silu
    _Float16* HU   = (_Float16*)carve((size_t)N * 64  * sizeof(_Float16)); // hidden of update MLP
    float*    cent = (float*)   carve((size_t)N * 4   * sizeof(float));
    u32*      pbW1 = (u32*)carve(4096 * sizeof(u32));   // K=64,  Nn=128
    u32*      pbW2 = (u32*)carve(2048 * sizeof(u32));   // K=64,  Nn=64
    u32*      pbWu1= (u32*)carve(4096 * sizeof(u32));   // K=128, Nn=64
    u32*      pbWu2= (u32*)carve(2048 * sizeof(u32));   // K=64,  Nn=64
    (void)ws_size; (void)n_in; (void)out_size;

    const int rowBlocks = N / 16;                // 3125, exact

    // prep + weight packing (independent)
    prep_kernel<<<(N * 64 + 255) / 256, 256, 0, stream>>>(feat, pos, U, cent, N);
    pack_b_kernel<<<16, 256, 0, stream>>>(W1,  pbW1, 64, 128, 1);
    pack_b_kernel<<< 8, 256, 0, stream>>>(W2,  pbW2, 64,  64, 0);
    pack_b_kernel<<<16, 256, 0, stream>>>(Wu1, pbWu1,128,  64, 0);
    pack_b_kernel<<< 8, 256, 0, stream>>>(Wu2, pbWu2, 64,  64, 0);

    // G1: X1 = f @ [W1a | W1b]   (N x 128, f32)
    gemm_wmma_kernel<2, 8, 128, 128, false, false, false, false>
        <<<rowBlocks, 256, 0, stream>>>(U, pbW1, nullptr, nullptr, X1);

    // pair pass: S = mean_j silu(A_i + B_j + g*w1g + b1)
    pair_pass_kernel<<<N / 4, 256, 0, stream>>>(X1, cent, nbr, b1, W1 + 128 * 64, S16, N);

    // G2: msg = S @ W2 + b2 -> f16 into U[:,64:128]
    gemm_wmma_kernel<2, 4, 64, 128, false, true, false, true>
        <<<rowBlocks, 128, 0, stream>>>(S16, pbW2, b2, nullptr, (void*)(U + 64));

    // G3: HU = silu([f | msg] @ Wu1 + bu1) -> f16   (K = 128)
    gemm_wmma_kernel<4, 4, 128, 64, true, true, false, true>
        <<<rowBlocks, 128, 0, stream>>>(U, pbWu1, bu1, nullptr, (void*)HU);

    // G4: out = feat + HU @ Wu2 + bu2   (f32, residual fused)
    gemm_wmma_kernel<2, 4, 64, 64, false, false, true, true>
        <<<rowBlocks, 128, 0, stream>>>(HU, pbWu2, bu2, feat, d_out);

    // positions pass through unchanged
    hipMemcpyAsync((float*)d_out + (size_t)N * 64, pos,
                   (size_t)4 * N * 3 * sizeof(float),
                   hipMemcpyDeviceToDevice, stream);
}
